// ConvClass_28295244546731
// MI455X (gfx1250) — matrix operand order repaired and checked by reading, not compiled, not run on hardware
//
#include <hip/hip_runtime.h>
#include <hip/hip_bf16.h>
#include <stdint.h>

// MI455X / gfx1250. wave32. WMMA 16x16 tiles.
// Pipeline:
//   k_prep : w0[512][256] = weight^T / sqrt(512*256)
//   25x { k_syrk: S = w^T w ; k_poly: M = 1.875I - 1.25S + 0.375 S^2 ; k_wm: w' = w M }
//   k_xw   : h[8192][256] = x @ w            (f32 WMMA 16x16x4, exact)
//   k_cvt  : ht[256][8192] = f16(h^T)        (K-contiguous for B fragments)
//   k_adj  : out = relu(adj @ h)             (f16 WMMA 16x16x32, f32 accum,
//                                             double-buffered LDS + async-to-LDS)
// adj (268 MB) streamed once -> memory-bound at ~11.5us on 23.3 TB/s; f16 WMMA
// keeps matrix throughput above the ~3 PFLOP/s the stream rate demands.

typedef __attribute__((ext_vector_type(2)))  float    v2f;
typedef __attribute__((ext_vector_type(8)))  float    v8f;
typedef __attribute__((ext_vector_type(4)))  _Float16 v4h;
typedef __attribute__((ext_vector_type(16))) _Float16 v16h;
typedef __attribute__((ext_vector_type(4)))  int      v4i;

typedef __attribute__((address_space(1))) v4i* gptr4;   // global v4i*
typedef __attribute__((address_space(3))) v4i* lptr4;   // LDS v4i*

#define NN   8192
#define DIN  512
#define DOUT 256

// ---------------- gfx1250 async global->LDS helpers ----------------
#if __has_builtin(__builtin_amdgcn_global_load_async_to_lds_b128)
#define HAVE_ASYNC_LDS 1
__device__ __forceinline__ void async_cp16(void* lds, const void* g) {
    __builtin_amdgcn_global_load_async_to_lds_b128(
        (gptr4)(g), (lptr4)(lds), 0, 0);
}
#else
#define HAVE_ASYNC_LDS 0
__device__ __forceinline__ void async_cp16(void* lds, const void* g) {
    *(uint4*)lds = *(const uint4*)g;          // sync fallback through VGPRs
}
#endif

template <int N>
__device__ __forceinline__ void wait_async() {
#if __has_builtin(__builtin_amdgcn_s_wait_asynccnt)
    __builtin_amdgcn_s_wait_asynccnt((unsigned short)N);
#elif HAVE_ASYNC_LDS
    asm volatile("s_wait_asynccnt %0" ::"n"(N) : "memory");
#endif
}

// ---------------- prep: w0 = weight^T / sqrt(131072) ----------------
__global__ void k_prep(const float* __restrict__ weight, float* __restrict__ w) {
    int idx = blockIdx.x * 256 + threadIdx.x;          // over 512*256
    int i = idx >> 8;                                  // input dim 0..511
    int o = idx & 255;                                 // output dim 0..255
    const float inv_s = 0.0027621358640484226f;        // 1/sqrt(131072)
    w[i * DOUT + o] = weight[o * DIN + i] * inv_s;
}

// ---------------- S = w^T w  (256x256, K=512) ----------------
__global__ __launch_bounds__(32) void k_syrk(const float* __restrict__ w,
                                             float* __restrict__ s) {
    int lane = threadIdx.x;
    int q = lane & 15, kh = lane >> 4;
    int a0 = blockIdx.x * 16, b0 = blockIdx.y * 16;
    v8f c = {};
    for (int k = 0; k < DIN; k += 4) {
        v2f A, B;
        A.x = w[(k + 2 * kh + 0) * DOUT + a0 + q];     // A[m][k'] = w[k'][a0+m]
        A.y = w[(k + 2 * kh + 1) * DOUT + a0 + q];
        B.x = w[(k + 2 * kh + 0) * DOUT + b0 + q];     // B[k'][n] = w[k'][b0+n]
        B.y = w[(k + 2 * kh + 1) * DOUT + b0 + q];
        c = __builtin_amdgcn_wmma_f32_16x16x4_f32(false, A, false, B,
                                                  (short)0, c, false, false);
    }
#pragma unroll
    for (int r = 0; r < 8; ++r)
        s[(a0 + r + 8 * kh) * DOUT + b0 + q] = c[r];
}

// ---------------- M = 0.375*S@S - 1.25*S + 1.875*I ----------------
__global__ __launch_bounds__(32) void k_poly(const float* __restrict__ s,
                                             float* __restrict__ m) {
    int lane = threadIdx.x;
    int q = lane & 15, kh = lane >> 4;
    int a0 = blockIdx.x * 16, b0 = blockIdx.y * 16;
    v8f c = {};
    for (int k = 0; k < DOUT; k += 4) {
        v2f A, B;
        A.x = s[(a0 + q) * DOUT + k + 2 * kh + 0];
        A.y = s[(a0 + q) * DOUT + k + 2 * kh + 1];
        B.x = s[(k + 2 * kh + 0) * DOUT + b0 + q];
        B.y = s[(k + 2 * kh + 1) * DOUT + b0 + q];
        c = __builtin_amdgcn_wmma_f32_16x16x4_f32(false, A, false, B,
                                                  (short)0, c, false, false);
    }
#pragma unroll
    for (int r = 0; r < 8; ++r) {
        int row = a0 + r + 8 * kh, col = b0 + q;
        float v = 0.375f * c[r] - 1.25f * s[row * DOUT + col];
        if (row == col) v += 1.875f;
        m[row * DOUT + col] = v;
    }
}

// ---------------- w' = w @ M  (512x256, K=256) ----------------
__global__ __launch_bounds__(32) void k_wm(const float* __restrict__ w,
                                           const float* __restrict__ m,
                                           float* __restrict__ wn) {
    int lane = threadIdx.x;
    int q = lane & 15, kh = lane >> 4;
    int m0 = blockIdx.x * 16, n0 = blockIdx.y * 16;
    v8f c = {};
    for (int k = 0; k < DOUT; k += 4) {
        v2f A, B;
        A.x = w[(m0 + q) * DOUT + k + 2 * kh + 0];
        A.y = w[(m0 + q) * DOUT + k + 2 * kh + 1];
        B.x = m[(k + 2 * kh + 0) * DOUT + n0 + q];
        B.y = m[(k + 2 * kh + 1) * DOUT + n0 + q];
        c = __builtin_amdgcn_wmma_f32_16x16x4_f32(false, A, false, B,
                                                  (short)0, c, false, false);
    }
#pragma unroll
    for (int r = 0; r < 8; ++r)
        wn[(m0 + r + 8 * kh) * DOUT + n0 + q] = c[r];
}

// ---------------- h = x @ w  (8192x256, K=512), LDS-staged f32 WMMA ----------------
__global__ __launch_bounds__(512) void k_xw(const float* __restrict__ x,
                                            const float* __restrict__ w,
                                            float* __restrict__ h) {
    __shared__ float xs[16][36];        // 16 rows x 32 K (padded)
    __shared__ float ws[32][264];       // 32 K x 256 N (padded)
    int t = threadIdx.x;
    int wid = t >> 5, lane = t & 31;
    int q = lane & 15, kh = lane >> 4;
    int m0 = blockIdx.x * 16;
    int n0 = wid * 16;
    v8f c = {};
    for (int k0 = 0; k0 < DIN; k0 += 32) {
        __syncthreads();
        { int r = t >> 5, cc = t & 31;
          xs[r][cc] = x[(size_t)(m0 + r) * DIN + k0 + cc]; }
#pragma unroll
        for (int j = 0; j < 16; ++j) {
            int idx = t + 512 * j;
            int r = idx >> 8, cc = idx & 255;
            ws[r][cc] = w[(k0 + r) * DOUT + cc];
        }
        __syncthreads();
#pragma unroll
        for (int kk = 0; kk < 32; kk += 4) {
            v2f A, B;
            A.x = xs[q][kk + 2 * kh + 0];
            A.y = xs[q][kk + 2 * kh + 1];
            B.x = ws[kk + 2 * kh + 0][n0 + q];
            B.y = ws[kk + 2 * kh + 1][n0 + q];
            c = __builtin_amdgcn_wmma_f32_16x16x4_f32(false, A, false, B,
                                                      (short)0, c, false, false);
        }
    }
#pragma unroll
    for (int r = 0; r < 8; ++r)
        h[(size_t)(m0 + r + 8 * kh) * DOUT + n0 + q] = c[r];
}

// ---------------- ht[n][k] = f16(h[k][n]) ----------------
__global__ void k_cvt(const float* __restrict__ h, _Float16* __restrict__ ht) {
    int idx = blockIdx.x * 256 + threadIdx.x;          // over 256*8192 (ht layout)
    int n = idx >> 13;                                 // 0..255
    int r = idx & 8191;                                // 0..8191
    ht[idx] = (_Float16)h[(size_t)r * DOUT + n];       // write coalesced, read hits L2
}

// ---------------- out = relu(adj @ h) ----------------
// Block: 512 thr / 16 waves. M=32 rows, N=256 cols, K-chunk=64.
// Per chunk per wave: 1 B fragment reused by 2 row-tiles -> 4 WMMAs.
// ht tile (32 KB, already f16) prefetched into next LDS buffer via async-to-LDS;
// adj tile (8 KB f32) loaded b128 per thread, converted once, stored to LDS.
__global__ __launch_bounds__(512) void k_adj(const float* __restrict__ adj,
                                             const _Float16* __restrict__ ht,
                                             float* __restrict__ out) {
    __shared__ _Float16 as[2][32][72];   // 32 rows x 64 K (144B rows, 16B aligned)
    __shared__ _Float16 hs[2][256][72];  // 256 N  x 64 K
    int t = threadIdx.x;
    int wid = t >> 5, lane = t & 31;
    int q = lane & 15, kh = lane >> 4;
    int m0 = blockIdx.x * 32;
    int n0 = wid * 16;

    // adj tile: thread -> row ar, 4 consecutive K at ac
    int ar = t >> 4;                 // 0..31
    int ac = (t & 15) * 4;           // 0..60
    const float* adj_src = adj + (size_t)(m0 + ar) * NN + ac;
    // ht tile: thread -> col hn, 32 consecutive K (64B) at hk
    int hn = t >> 1;                 // 0..255
    int hk = (t & 1) * 32;           // 0 or 32
    const _Float16* ht_src = ht + (size_t)hn * NN + hk;

    // ---- prologue: prefetch chunk 0 ----
#pragma unroll
    for (int j = 0; j < 4; ++j)
        async_cp16(&hs[0][hn][hk + 8 * j], ht_src + 8 * j);
    float4 areg = *(const float4*)adj_src;

    v8f c0 = {}, c1 = {};
    for (int i = 0; i < NN / 64; ++i) {
        int buf = i & 1;
        // convert + store adj tile (last read of as[buf] was in iter i-2)
        v4h hv;
        hv[0] = (_Float16)areg.x; hv[1] = (_Float16)areg.y;
        hv[2] = (_Float16)areg.z; hv[3] = (_Float16)areg.w;
        *(v4h*)&as[buf][ar][ac] = hv;

        if (i + 1 < NN / 64) {
            // prefetch chunk i+1 into the other buffer
#pragma unroll
            for (int j = 0; j < 4; ++j)
                async_cp16(&hs[buf ^ 1][hn][hk + 8 * j],
                           ht_src + (i + 1) * 64 + 8 * j);
            areg = *(const float4*)(adj_src + (i + 1) * 64);
            wait_async<4>();         // chunk i's 4 copies retired
        } else {
            wait_async<0>();
        }
        __syncthreads();

#pragma unroll
        for (int ks = 0; ks < 64; ks += 32) {
            v16h B;
#pragma unroll
            for (int e = 0; e < 16; ++e)            // B col n0+q, K = ks + 8*kh + (e&7) + 16*(e>>3)
                B[e] = hs[buf][n0 + q][ks + 8 * kh + (e & 7) + 16 * (e >> 3)];
            v16h A0, A1;
#pragma unroll
            for (int e = 0; e < 8; ++e) {           // A rows: VGPR0-3 K=8kh+e, VGPR4-7 K=16+8kh+e
                A0[e]     = as[buf][q][ks + 8 * kh + e];
                A0[8 + e] = as[buf][q][ks + 16 + 8 * kh + e];
                A1[e]     = as[buf][16 + q][ks + 8 * kh + e];
                A1[8 + e] = as[buf][16 + q][ks + 16 + 8 * kh + e];
            }
            c0 = __builtin_amdgcn_wmma_f32_16x16x32_f16(false, A0, false, B,
                                                        (short)0, c0, false, false);
            c1 = __builtin_amdgcn_wmma_f32_16x16x32_f16(false, A1, false, B,
                                                        (short)0, c1, false, false);
        }
        __syncthreads();
    }

#pragma unroll
    for (int r = 0; r < 8; ++r) {
        float v0 = c0[r], v1 = c1[r];
        out[(size_t)(m0 + r + 8 * kh) * DOUT + n0 + q]      = v0 > 0.0f ? v0 : 0.0f;
        out[(size_t)(m0 + 16 + r + 8 * kh) * DOUT + n0 + q] = v1 > 0.0f ? v1 : 0.0f;
    }
}

extern "C" void kernel_launch(void* const* d_in, const int* in_sizes, int n_in,
                              void* d_out, int out_size, void* d_ws, size_t ws_size,
                              hipStream_t stream) {
    const float* x      = (const float*)d_in[0];   // [8192,512]
    const float* adj    = (const float*)d_in[1];   // [8192,8192]
    const float* weight = (const float*)d_in[2];   // [256,512]
    float* out = (float*)d_out;                    // [8192,256]

    char* ws = (char*)d_ws;
    float*    w_a = (float*)(ws);
    float*    w_b = (float*)(ws + (512ull << 10));
    float*    s   = (float*)(ws + (1024ull << 10));
    float*    m   = (float*)(ws + (1280ull << 10));
    float*    h   = (float*)(ws + (1536ull << 10));
    _Float16* ht  = (_Float16*)(ws + ((1536ull + 8192ull) << 10));

    k_prep<<<(DIN * DOUT) / 256, 256, 0, stream>>>(weight, w_a);

    float* cur = w_a;
    float* nxt = w_b;
    for (int it = 0; it < 25; ++it) {
        k_syrk<<<dim3(16, 16), 32, 0, stream>>>(cur, s);
        k_poly<<<dim3(16, 16), 32, 0, stream>>>(s, m);
        k_wm<<<dim3(32, 16), 32, 0, stream>>>(cur, m, nxt);
        float* tmp = cur; cur = nxt; nxt = tmp;
    }

    k_xw<<<NN / 16, 512, 0, stream>>>(x, cur, h);
    k_cvt<<<(NN * DOUT) / 256, 256, 0, stream>>>(h, ht);
    k_adj<<<NN / 32, 512, 0, stream>>>(adj, ht, out);
}
